// SpectrogramModel_75668733821468
// MI455X (gfx1250) — matrix operand order repaired and checked by reading, not compiled, not run on hardware
//
#include <hip/hip_runtime.h>
#include <stdint.h>
#include <stddef.h>

typedef __bf16 bf16;
typedef __attribute__((ext_vector_type(16))) __bf16 v16bf;
typedef __attribute__((ext_vector_type(8)))  __bf16 v8bf;
typedef __attribute__((ext_vector_type(8)))  float  v8f;

#define DEVI static __device__ __forceinline__

DEVI float bf2f(bf16 x){ return (float)x; }
DEVI bf16  f2bf(float x){ return (bf16)x; }
DEVI float sigf(float x){ return 1.0f/(1.0f+__expf(-x)); }

DEVI v8f zero8(){
  v8f r;
  #pragma unroll
  for(int i=0;i<8;++i) r[i]=0.0f;
  return r;
}

DEVI v8f wmma_bf(v16bf a, v16bf b, v8f c){
  // D = A(16x32 bf16) * B(32x16 bf16) + C(f32)
  return __builtin_amdgcn_wmma_f32_16x16x32_bf16(false, a, false, b, (short)0, c, false, false);
}

// A fragment: 16x32 bf16 tile, row-major with row stride ld (rows = M).
// Each lane's 16 values are one contiguous 16-element run of its row
// (k = 16*(lane>>4) + s), matching the swizzled B -> two b128 loads.
DEVI v16bf load_a_ld(const bf16* a, int ld, int lane){
  const int m  = lane & 15;
  const int hi = (lane >> 4) ? 16 : 0;
  const bf16* row = a + (size_t)m*ld + hi;
  v8bf lo = *(const v8bf*)(row);
  v8bf hh = *(const v8bf*)(row + 8);
  return __builtin_shufflevector(lo, hh, 0,1,2,3,4,5,6,7,8,9,10,11,12,13,14,15);
}

// B fragment from pre-swizzled weights: layout [K/32][N/16][lane32][16],
// each lane's 16 values contiguous (32B) -> two b128 loads.
DEVI v16bf load_b_sw(const bf16* Wsw, size_t tile, int lane){
  const bf16* ptr = Wsw + tile*512 + lane*16;
  v8bf lo = *(const v8bf*)(ptr);
  v8bf hh = *(const v8bf*)(ptr + 8);
  return __builtin_shufflevector(lo, hh, 0,1,2,3,4,5,6,7,8,9,10,11,12,13,14,15);
}
// C fragment element (m,n) for lane l, index v: n = l&15, m = v + 8*(l>>4).

// ---------------------------------------------------------------- utility kernels

__global__ void zero_bf16_k(bf16* p, size_t n){
  size_t i = (size_t)blockIdx.x*blockDim.x + threadIdx.x;
  if(i < n) p[i] = (bf16)0.0f;
}

// Swizzled B-weight prep: dst[Kp/32][N/16][32][16] <- src[N][K] (transposed),
// zero pad k>=K. If H>0: dst column n maps to src row (n&3)*H + (n>>2)
// (torch i,f,g,o gate interleave so one 16-wide tile holds 4 complete units).
__global__ void prep_bw_k(const float* __restrict__ src, bf16* __restrict__ dst,
                          int N, int K, int Kp, int H){
  size_t i = (size_t)blockIdx.x*blockDim.x + threadIdx.x;
  if(i >= (size_t)Kp*N) return;
  int NT = N >> 4;
  int j    = (int)(i & 15);
  int lane = (int)((i >> 4) & 31);
  size_t tile = i >> 9;
  int kt = (int)(tile / NT);
  int nt = (int)(tile % NT);
  int n = nt*16 + (lane & 15);
  int k = kt*32 + ((lane >> 4) ? 16 : 0) + j;
  int ns = H ? ((n & 3)*H + (n >> 2)) : n;
  dst[i] = (k < K) ? f2bf(src[(size_t)ns*K + k]) : (bf16)0.0f;
}

__global__ void interleave_bias_k(const float* __restrict__ src, float* __restrict__ dst, int H){
  int i = blockIdx.x*blockDim.x + threadIdx.x;
  if(i >= 4*H) return;
  dst[i] = src[(i & 3)*H + (i >> 2)];
}

// Swizzled conv-weight prep: dst[taps][Cip/32][Co/16][32][16] <- src[Co][Ci][taps]
__global__ void prep_convw_k(const float* __restrict__ src, bf16* __restrict__ dst,
                             int Co, int Ci, int taps, int Cip){
  size_t i = (size_t)blockIdx.x*blockDim.x + threadIdx.x;
  size_t total = (size_t)taps*Cip*Co;
  if(i >= total) return;
  int NT = Co >> 4, KT = Cip >> 5;
  int j    = (int)(i & 15);
  int lane = (int)((i >> 4) & 31);
  size_t tile = i >> 9;
  int nt  = (int)(tile % NT);
  int kt  = (int)((tile / NT) % KT);
  int tap = (int)(tile / ((size_t)NT*KT));
  int o = nt*16 + (lane & 15);
  int c = kt*32 + ((lane >> 4) ? 16 : 0) + j;
  dst[i] = (c < Ci) ? f2bf(src[((size_t)o*Ci + c)*taps + tap]) : (bf16)0.0f;
}

// embedding -> padded [8][132][512] bf16 (halo slots pre-zeroed)
__global__ void embed_k(const int* __restrict__ tokens, const float* __restrict__ emb,
                        bf16* __restrict__ out){
  size_t i = (size_t)blockIdx.x*blockDim.x + threadIdx.x;
  if(i >= (size_t)8*128*512) return;
  int c = (int)(i & 511);
  int t = (int)((i >> 9) & 127);
  int b = (int)(i >> 16);
  int tok = tokens[t*8 + b];
  out[((size_t)b*132 + t + 2)*512 + c] = f2bf(emb[(size_t)tok*512 + c]);
}

// frames [200][8][80] f32 -> pnIn [8][212][96] bf16 (slot t+2, channel pad)
__global__ void pn_in_k(const float* __restrict__ frames, bf16* __restrict__ pnIn){
  size_t i = (size_t)blockIdx.x*blockDim.x + threadIdx.x;
  if(i >= (size_t)8*200*96) return;
  int c = (int)(i % 96);
  int t = (int)((i / 96) % 200);
  int b = (int)(i / (96*200));
  float v = (c < 80) ? frames[((size_t)t*8 + b)*80 + c] : 0.0f;
  pnIn[((size_t)b*212 + t + 2)*96 + c] = f2bf(v);
}

// mel2[t][b][c] = frames[t][b][c] + resid[b][t][c]
__global__ void combine_k(const float* __restrict__ frames, const float* __restrict__ resid,
                          float* __restrict__ out2){
  int i = blockIdx.x*blockDim.x + threadIdx.x;
  if(i >= 128000) return;
  int c = i % 80;
  int b = (i / 80) % 8;
  int t = i / 640;
  out2[i] = frames[i] + resid[((size_t)b*200 + t)*80 + c];
}

// ---------------------------------------------------------------- generic conv/GEMM (1 wave/block)
struct GemmP {
  const bf16* A; const bf16* W; const float* bias;
  const float* bn_g; const float* bn_b; const float* bn_m; const float* bn_v;
  bf16* outB; float* outF;
  int Bsz, Trows, TT, taps, Cip, Co, TpadIn, tOffIn, TpadOut, tOffOut, ldcF, act;
};

__global__ __launch_bounds__(32) void gemm_conv_k(GemmP p){
  int lane = threadIdx.x;
  int bi = blockIdx.x / p.TT;
  int t0 = (blockIdx.x % p.TT) * 16;
  int nt = blockIdx.y;
  int n0 = nt * 16;
  int NT = p.Co >> 4, KT = p.Cip >> 5;
  v8f acc = zero8();
  for(int tap=0; tap<p.taps; ++tap){
    const bf16* Abase = p.A + ((size_t)bi*p.TpadIn + t0 + tap + p.tOffIn)*p.Cip;
    for(int kc=0; kc<p.Cip; kc+=32){
      v16bf af = load_a_ld(Abase + kc, p.Cip, lane);
      size_t tile = (size_t)(tap*KT + (kc >> 5))*NT + nt;
      v16bf bw = load_b_sw(p.W, tile, lane);
      acc = wmma_bf(af, bw, acc);
    }
  }
  int n  = n0 + (lane & 15);
  int mb = (lane >> 4) << 3;
  #pragma unroll
  for(int v=0; v<8; ++v){
    int t = t0 + mb + v;
    float y = acc[v];
    if(p.bias) y += p.bias[n];
    if(p.bn_g) y = p.bn_g[n]*(y - p.bn_m[n])*rsqrtf(p.bn_v[n] + 1e-5f) + p.bn_b[n];
    if(p.act == 1) y = fmaxf(y, 0.0f);
    else if(p.act == 2) y = tanhf(y);
    if(t < p.Trows){
      if(p.outB) p.outB[((size_t)bi*p.TpadOut + t + p.tOffOut)*p.Co + n] = f2bf(y);
      if(p.outF) p.outF[((size_t)bi*p.Trows + t)*p.ldcF + n] = y;
    }
  }
}

// ---------------------------------------------------------------- BiLSTM scan (2 blocks x 256 thr)
struct LstmScanP {
  const float* xW_f; const float* xW_b;   // [8][128][1024] gate-interleaved, bias included
  const bf16* WhhT_f; const bf16* WhhT_b; // swizzled [256/32][1024/16][32][16]
  float* memF;                            // [8][128][512]
  bf16*  memB;                            // [8][128][512]
};

__global__ __launch_bounds__(256) void lstm_scan_k(LstmScanP p){
  __shared__ alignas(16) bf16  hpad[16*256];
  __shared__ float scr[8][256];
  int tid = threadIdx.x, lane = tid & 31, wave = tid >> 5;
  int dir = blockIdx.x;
  const float* xW  = dir ? p.xW_b  : p.xW_f;
  const bf16*  Whh = dir ? p.WhhT_b : p.WhhT_f;
  for(int i=tid; i<16*256; i+=256) hpad[i] = (bf16)0.0f;
  float cr[8];
  #pragma unroll
  for(int i=0;i<8;++i) cr[i]=0.0f;
  __syncthreads();
  for(int s=0; s<128; ++s){
    int t = dir ? (127 - s) : s;
    v8f acc[8];
    #pragma unroll
    for(int i=0;i<8;++i) acc[i]=zero8();
    for(int kc=0; kc<256; kc+=32){
      v16bf af = load_a_ld(hpad + kc, 256, lane);
      #pragma unroll
      for(int i=0;i<8;++i){
        v16bf bw = load_b_sw(Whh, (size_t)(kc>>5)*64 + (wave + 8*i), lane);
        acc[i] = wmma_bf(af, bw, acc[i]);
      }
    }
    __syncthreads();            // all reads of hpad complete
    #pragma unroll
    for(int i=0;i<8;++i){
      int n0 = (wave + 8*i)*16;
      { int nn = lane & 15, mb = (lane >> 4) << 3;
        #pragma unroll
        for(int v=0;v<8;++v) scr[wave][(mb+v)*16 + nn] = acc[i][v]; }
      int b = lane & 7, up = lane >> 3, c4 = 4*up;
      const float* xrow = &xW[((size_t)b*128 + t)*1024];
      float zi = scr[wave][b*16 + c4 + 0] + xrow[n0 + c4 + 0];
      float zf = scr[wave][b*16 + c4 + 1] + xrow[n0 + c4 + 1];
      float zg = scr[wave][b*16 + c4 + 2] + xrow[n0 + c4 + 2];
      float zo = scr[wave][b*16 + c4 + 3] + xrow[n0 + c4 + 3];
      float cn = sigf(zf)*cr[i] + sigf(zi)*tanhf(zg);
      float hn = sigf(zo)*tanhf(cn);
      cr[i] = cn;
      int u = (n0 >> 2) + up;
      hpad[b*256 + u] = f2bf(hn);
      size_t mi = ((size_t)b*128 + t)*512 + dir*256 + u;
      p.memF[mi] = hn;
      p.memB[mi] = f2bf(hn);
    }
    __syncthreads();
  }
}

// ---------------------------------------------------------------- persistent decoder (1 block x 1024)
struct DecP {
  const float* gt;        // [200][8][80]
  const bf16* preW1T;     // swizzled [96/32][256/16][32][16]
  const float* pre_b1;
  const bf16* preW2T;     // swizzled K=256,N=256
  const float* pre_b2;
  const bf16* d1WihT;     // swizzled K=768 (x256|ctx512), N=4096 gate-interleaved
  const bf16* d1WhhT;     // swizzled K=1024, N=4096
  const float* d1bI;      // [4096]
  const bf16* d2WihT;     // swizzled K=1536 (h1|ctx), N=4096
  const bf16* d2WhhT;     // swizzled K=1024, N=4096
  const float* d2bI;      // [4096]
  const bf16* attWqT;     // swizzled K=1024, N=128
  const float* attWl;     // [128][32]
  const float* attLocW;   // [32][2][31]
  const float* attV;      // [128]
  const float* procMem;   // [8][128][128]
  const float* memF;      // [8][128][512]
  const int*   tokens;    // [128][8]
  const bf16* projT;      // swizzled K=1536, N=80
  const float* proj_b;    // [80]
  const float* stopW;     // [1536]
  const float* stop_b;    // [1]
  float* frames;          // [200][8][80]
  float* stops;           // [200][8]
  float* aligns;          // [200][8][128]
};

__global__ __launch_bounds__(1024) void decoder_k(DecP p){
  __shared__ alignas(16) bf16  h1pad[16*1024];
  __shared__ alignas(16) bf16  h2pad[16*1024];
  __shared__ alignas(16) bf16  ctxpad[16*512];
  __shared__ alignas(16) bf16  x1pad[16*256];
  __shared__ alignas(16) bf16  x2pad[16*256];
  __shared__ alignas(16) bf16  prevpad[16*96];
  __shared__ float scr[32][256];   // per-wave gate scratch; also reused as Wl cache [128][32]
  __shared__ float qbuf[8*128];
  __shared__ float ebuf[8*128];
  __shared__ float abuf[8*128];
  __shared__ float acum[8*128];
  __shared__ float locv[2112];     // [32][2][31] loc conv weights + v[128]

  int tid = threadIdx.x, lane = tid & 31, wave = tid >> 5;
  for(int i=tid; i<16*1024; i+=1024){ h1pad[i]=(bf16)0.0f; h2pad[i]=(bf16)0.0f; }
  for(int i=tid; i<16*512;  i+=1024) ctxpad[i]=(bf16)0.0f;
  for(int i=tid; i<16*256;  i+=1024){ x1pad[i]=(bf16)0.0f; x2pad[i]=(bf16)0.0f; }
  for(int i=tid; i<16*96;   i+=1024) prevpad[i]=(bf16)0.0f;
  if(tid < 1024){ abuf[tid]=0.0f; acum[tid]=0.0f; }
  for(int i=tid; i<1984; i+=1024) locv[i] = p.attLocW[i];
  if(tid < 128) locv[1984 + tid] = p.attV[tid];
  float c1r[8], c2r[8];
  #pragma unroll
  for(int i=0;i<8;++i){ c1r[i]=0.0f; c2r[i]=0.0f; }
  __syncthreads();

  for(int t=0; t<200; ++t){
    // ---- phase A: previous frame into prenet pad
    if(tid < 8*96){
      int b = tid / 96, c = tid % 96;
      float v = 0.0f;
      if(c < 80 && t > 0) v = p.gt[((size_t)(t-1)*8 + b)*80 + c];
      prevpad[b*96 + c] = f2bf(v);
    }
    __syncthreads();
    // ---- phase B: prenet layer 1 (waves 0..15)
    if(wave < 16){
      v8f acc = zero8();
      for(int kc=0; kc<96; kc+=32){
        v16bf af = load_a_ld(prevpad + kc, 96, lane);
        v16bf bw = load_b_sw(p.preW1T, (size_t)(kc>>5)*16 + wave, lane);
        acc = wmma_bf(af, bw, acc);
      }
      if(lane < 16){ int n = wave*16 + lane;
        #pragma unroll
        for(int v=0;v<8;++v){ float y = fmaxf(acc[v] + p.pre_b1[n], 0.0f); x1pad[v*256 + n] = f2bf(y); } }
    }
    __syncthreads();
    // ---- phase C: prenet layer 2
    if(wave < 16){
      v8f acc = zero8();
      for(int kc=0; kc<256; kc+=32){
        v16bf af = load_a_ld(x1pad + kc, 256, lane);
        v16bf bw = load_b_sw(p.preW2T, (size_t)(kc>>5)*16 + wave, lane);
        acc = wmma_bf(af, bw, acc);
      }
      if(lane < 16){ int n = wave*16 + lane;
        #pragma unroll
        for(int v=0;v<8;++v){ float y = fmaxf(acc[v] + p.pre_b2[n], 0.0f); x2pad[v*256 + n] = f2bf(y); } }
    }
    __syncthreads();
    // ---- phase D: attention LSTM (dec1), all 32 waves x 8 tiles of N=4096
    {
      v8f acc[8];
      #pragma unroll
      for(int i=0;i<8;++i) acc[i]=zero8();
      for(int kc=0; kc<256; kc+=32){
        v16bf af = load_a_ld(x2pad + kc, 256, lane);
        #pragma unroll
        for(int i=0;i<8;++i){
          v16bf bw = load_b_sw(p.d1WihT, (size_t)(kc>>5)*256 + (wave + 32*i), lane);
          acc[i] = wmma_bf(af, bw, acc[i]);
        }
      }
      for(int kc=0; kc<512; kc+=32){
        v16bf af = load_a_ld(ctxpad + kc, 512, lane);
        #pragma unroll
        for(int i=0;i<8;++i){
          v16bf bw = load_b_sw(p.d1WihT, (size_t)(8 + (kc>>5))*256 + (wave + 32*i), lane);
          acc[i] = wmma_bf(af, bw, acc[i]);
        }
      }
      for(int kc=0; kc<1024; kc+=32){
        v16bf af = load_a_ld(h1pad + kc, 1024, lane);
        #pragma unroll
        for(int i=0;i<8;++i){
          v16bf bw = load_b_sw(p.d1WhhT, (size_t)(kc>>5)*256 + (wave + 32*i), lane);
          acc[i] = wmma_bf(af, bw, acc[i]);
        }
      }
      __syncthreads();  // all reads of h1pad/ctxpad done before h1 update
      #pragma unroll
      for(int i=0;i<8;++i){
        int n0 = (wave + 32*i)*16;
        { int nn = lane & 15, mb = (lane >> 4) << 3;
          #pragma unroll
          for(int v=0;v<8;++v) scr[wave][(mb+v)*16 + nn] = acc[i][v]; }
        int b = lane & 7, up = lane >> 3, c4 = 4*up;
        float zi = scr[wave][b*16 + c4 + 0] + p.d1bI[n0 + c4 + 0];
        float zf = scr[wave][b*16 + c4 + 1] + p.d1bI[n0 + c4 + 1];
        float zg = scr[wave][b*16 + c4 + 2] + p.d1bI[n0 + c4 + 2];
        float zo = scr[wave][b*16 + c4 + 3] + p.d1bI[n0 + c4 + 3];
        float cn = sigf(zf)*c1r[i] + sigf(zi)*tanhf(zg);
        float hn = sigf(zo)*tanhf(cn);
        c1r[i] = cn;
        h1pad[b*1024 + (n0 >> 2) + up] = f2bf(hn);
      }
    }
    __syncthreads();
    // ---- phase E1: query projection q = h1 @ WqT (waves 0..7);
    //      waves 8..31 concurrently stage att_Wl into scr (reused as [128][32])
    if(wave < 8){
      v8f acc = zero8();
      for(int kc=0; kc<1024; kc+=32){
        v16bf af = load_a_ld(h1pad + kc, 1024, lane);
        v16bf bw = load_b_sw(p.attWqT, (size_t)(kc>>5)*8 + wave, lane);
        acc = wmma_bf(af, bw, acc);
      }
      if(lane < 16){ int n = wave*16 + lane;
        #pragma unroll
        for(int v=0;v<8;++v) qbuf[v*128 + n] = acc[v]; }
    } else {
      float* wl = &scr[0][0];
      for(int i = tid - 256; i < 4096; i += 768) wl[i] = p.attWl[i];
    }
    __syncthreads();
    // ---- phase E2: location features + energies (one thread per (b,t))
    {
      const float* wl = &scr[0][0];
      int b = tid >> 7, tt = tid & 127;
      float feat[32];
      #pragma unroll
      for(int f=0;f<32;++f) feat[f]=0.0f;
      for(int j=0;j<31;++j){
        int tj = tt + j - 15;
        if(tj >= 0 && tj < 128){
          float ap = abuf[b*128 + tj], ac = acum[b*128 + tj];
          #pragma unroll
          for(int f=0;f<32;++f)
            feat[f] += ap*locv[f*62 + j] + ac*locv[f*62 + 31 + j];
        }
      }
      float e = 0.0f;
      for(int d=0; d<128; ++d){
        float lp = 0.0f;
        #pragma unroll
        for(int f=0;f<32;++f) lp += feat[f]*wl[d*32 + f];
        float s = tanhf(qbuf[b*128 + d] + p.procMem[((size_t)b*128 + tt)*128 + d] + lp);
        e += s*locv[1984 + d];
      }
      if(p.tokens[tt*8 + b] == 0) e = -1e9f;
      ebuf[b*128 + tt] = e;
    }
    __syncthreads();
    // ---- phase E3: softmax over T (wave b handles batch row b)
    if(wave < 8){
      int b = wave;
      float ev[4], m0 = -1e30f;
      #pragma unroll
      for(int q=0;q<4;++q){ ev[q] = ebuf[b*128 + lane + 32*q]; m0 = fmaxf(m0, ev[q]); }
      #pragma unroll
      for(int off=16; off>0; off>>=1) m0 = fmaxf(m0, __shfl_xor(m0, off));
      float s = 0.0f;
      #pragma unroll
      for(int q=0;q<4;++q){ ev[q] = __expf(ev[q] - m0); s += ev[q]; }
      #pragma unroll
      for(int off=16; off>0; off>>=1) s += __shfl_xor(s, off);
      float inv = 1.0f / s;
      #pragma unroll
      for(int q=0;q<4;++q){
        int tt = lane + 32*q;
        float a = ev[q]*inv;
        abuf[b*128 + tt] = a;
        acum[b*128 + tt] += a;
        p.aligns[((size_t)t*8 + b)*128 + tt] = a;
      }
    }
    __syncthreads();
    // ---- phase E4: context = a @ memory   (4 outputs per thread)
    {
      int o0 = tid*4;
      int b = o0 >> 9, d0 = o0 & 511;
      float s0=0.f,s1=0.f,s2=0.f,s3=0.f;
      for(int tt=0; tt<128; ++tt){
        float a = abuf[b*128 + tt];
        const float* mrow = &p.memF[((size_t)b*128 + tt)*512 + d0];
        s0 += a*mrow[0]; s1 += a*mrow[1]; s2 += a*mrow[2]; s3 += a*mrow[3];
      }
      ctxpad[b*512 + d0 + 0] = f2bf(s0);
      ctxpad[b*512 + d0 + 1] = f2bf(s1);
      ctxpad[b*512 + d0 + 2] = f2bf(s2);
      ctxpad[b*512 + d0 + 3] = f2bf(s3);
    }
    __syncthreads();
    // ---- phase F: decoder LSTM (dec2)
    {
      v8f acc[8];
      #pragma unroll
      for(int i=0;i<8;++i) acc[i]=zero8();
      for(int kc=0; kc<1024; kc+=32){
        v16bf af = load_a_ld(h1pad + kc, 1024, lane);
        #pragma unroll
        for(int i=0;i<8;++i){
          v16bf bw = load_b_sw(p.d2WihT, (size_t)(kc>>5)*256 + (wave + 32*i), lane);
          acc[i] = wmma_bf(af, bw, acc[i]);
        }
      }
      for(int kc=0; kc<512; kc+=32){
        v16bf af = load_a_ld(ctxpad + kc, 512, lane);
        #pragma unroll
        for(int i=0;i<8;++i){
          v16bf bw = load_b_sw(p.d2WihT, (size_t)(32 + (kc>>5))*256 + (wave + 32*i), lane);
          acc[i] = wmma_bf(af, bw, acc[i]);
        }
      }
      for(int kc=0; kc<1024; kc+=32){
        v16bf af = load_a_ld(h2pad + kc, 1024, lane);
        #pragma unroll
        for(int i=0;i<8;++i){
          v16bf bw = load_b_sw(p.d2WhhT, (size_t)(kc>>5)*256 + (wave + 32*i), lane);
          acc[i] = wmma_bf(af, bw, acc[i]);
        }
      }
      __syncthreads();
      #pragma unroll
      for(int i=0;i<8;++i){
        int n0 = (wave + 32*i)*16;
        { int nn = lane & 15, mb = (lane >> 4) << 3;
          #pragma unroll
          for(int v=0;v<8;++v) scr[wave][(mb+v)*16 + nn] = acc[i][v]; }
        int b = lane & 7, up = lane >> 3, c4 = 4*up;
        float zi = scr[wave][b*16 + c4 + 0] + p.d2bI[n0 + c4 + 0];
        float zf = scr[wave][b*16 + c4 + 1] + p.d2bI[n0 + c4 + 1];
        float zg = scr[wave][b*16 + c4 + 2] + p.d2bI[n0 + c4 + 2];
        float zo = scr[wave][b*16 + c4 + 3] + p.d2bI[n0 + c4 + 3];
        float cn = sigf(zf)*c2r[i] + sigf(zi)*tanhf(zg);
        float hn = sigf(zo)*tanhf(cn);
        c2r[i] = cn;
        h2pad[b*1024 + (n0 >> 2) + up] = f2bf(hn);
      }
    }
    __syncthreads();
    // ---- phase G: frame projection (waves 0..4, N=80) and stop gate (wave 5)
    if(wave < 5){
      v8f acc = zero8();
      for(int kc=0; kc<1024; kc+=32){
        v16bf af = load_a_ld(h2pad + kc, 1024, lane);
        v16bf bw = load_b_sw(p.projT, (size_t)(kc>>5)*5 + wave, lane);
        acc = wmma_bf(af, bw, acc);
      }
      for(int kc=0; kc<512; kc+=32){
        v16bf af = load_a_ld(ctxpad + kc, 512, lane);
        v16bf bw = load_b_sw(p.projT, (size_t)(32 + (kc>>5))*5 + wave, lane);
        acc = wmma_bf(af, bw, acc);
      }
      if(lane < 16){ int n = wave*16 + lane;
        #pragma unroll
        for(int v=0;v<8;++v) p.frames[((size_t)t*8 + v)*80 + n] = acc[v] + p.proj_b[n]; }
    }
    if(wave == 5){
      for(int b=0;b<8;++b){
        float sacc = 0.0f;
        for(int k=lane; k<1536; k+=32){
          float hv = (k < 1024) ? bf2f(h2pad[b*1024 + k]) : bf2f(ctxpad[b*512 + k - 1024]);
          sacc += hv * p.stopW[k];
        }
        #pragma unroll
        for(int off=16; off>0; off>>=1) sacc += __shfl_xor(sacc, off);
        if(lane == 0) p.stops[t*8 + b] = 1.0f/(1.0f + __expf(-(sacc + p.stop_b[0])));
      }
    }
    __syncthreads();
  }
}

// ---------------------------------------------------------------- host driver

extern "C" void kernel_launch(void* const* d_in, const int* in_sizes, int n_in,
                              void* d_out, int out_size, void* d_ws, size_t ws_size,
                              hipStream_t stream){
  (void)in_sizes; (void)n_in; (void)out_size; (void)ws_size;
  const int*   tokens = (const int*)  d_in[0];
  const float* gt     = (const float*)d_in[1];
  const float* emb    = (const float*)d_in[2];
  auto FP = [&](int i){ return (const float*)d_in[i]; };

  // ---- workspace layout (cursor allocation; all counts keep 16B alignment) ----
  char* cur = (char*)d_ws;
  auto takeB = [&](size_t n)->bf16*{ bf16* r=(bf16*)cur; cur += n*sizeof(bf16); return r; };
  auto takeF = [&](size_t n)->float*{ float* r=(float*)cur; cur += n*sizeof(float); return r; };

  // contiguous zero-initialized activation block (halo/pad regions must be 0)
  bf16* embPad = takeB((size_t)8*132*512);
  bf16* cbufA  = takeB((size_t)8*132*512);
  bf16* cbufB  = takeB((size_t)8*132*512);
  bf16* pnIn   = takeB((size_t)8*212*96);
  bf16* pnB0   = takeB((size_t)8*212*512);
  bf16* pnB1   = takeB((size_t)8*212*512);
  size_t zeroN = (size_t)8*132*512*3 + (size_t)8*212*96 + (size_t)8*212*512*2;

  bf16* encWT[3]; for(int i=0;i<3;++i) encWT[i] = takeB((size_t)5*512*512);
  bf16* WihT_f = takeB((size_t)512*1024);
  bf16* WihT_b = takeB((size_t)512*1024);
  bf16* WhhT_f = takeB((size_t)256*1024);
  bf16* WhhT_b = takeB((size_t)256*1024);
  bf16* attWqT = takeB((size_t)1024*128);
  bf16* attWmT = takeB((size_t)512*128);
  bf16* preW1T = takeB((size_t)96*256);
  bf16* preW2T = takeB((size_t)256*256);
  bf16* d1WihT = takeB((size_t)768*4096);
  bf16* d1WhhT = takeB((size_t)1024*4096);
  bf16* d2WihT = takeB((size_t)1536*4096);
  bf16* d2WhhT = takeB((size_t)1024*4096);
  bf16* projT  = takeB((size_t)1536*80);
  bf16* pnWT[5];
  pnWT[0] = takeB((size_t)5*96*512);
  for(int i=1;i<4;++i) pnWT[i] = takeB((size_t)5*512*512);
  pnWT[4] = takeB((size_t)5*512*80);
  bf16* memB = takeB((size_t)8*128*512);

  float* lstmBI_f = takeF(1024);
  float* lstmBI_b = takeF(1024);
  float* d1bI = takeF(4096);
  float* d2bI = takeF(4096);
  float* xW_f = takeF((size_t)8*128*1024);
  float* xW_b = takeF((size_t)8*128*1024);
  float* memF = takeF((size_t)8*128*512);
  float* procMem = takeF((size_t)8*128*128);
  float* resid = takeF((size_t)8*200*80);

  auto G = [&](size_t n){ return dim3((unsigned)((n + 255)/256)); };

  // ---- weight preparation (f32 -> bf16, transpose, gate interleave, WMMA swizzle) ----
  zero_bf16_k<<<G(zeroN),256,0,stream>>>(embPad, zeroN);
  for(int i=0;i<3;++i)
    prep_convw_k<<<G((size_t)5*512*512),256,0,stream>>>(FP(3+6*i), encWT[i], 512,512,5,512);
  prep_bw_k<<<G((size_t)512*1024),256,0,stream>>>(FP(21), WihT_f, 1024,512,512,256);
  prep_bw_k<<<G((size_t)256*1024),256,0,stream>>>(FP(22), WhhT_f, 1024,256,256,256);
  interleave_bias_k<<<G(1024),256,0,stream>>>(FP(23), lstmBI_f, 256);
  prep_bw_k<<<G((size_t)512*1024),256,0,stream>>>(FP(24), WihT_b, 1024,512,512,256);
  prep_bw_k<<<G((size_t)256*1024),256,0,stream>>>(FP(25), WhhT_b, 1024,256,256,256);
  interleave_bias_k<<<G(1024),256,0,stream>>>(FP(26), lstmBI_b, 256);
  prep_bw_k<<<G((size_t)1024*128),256,0,stream>>>(FP(27), attWqT, 128,1024,1024,0);
  prep_bw_k<<<G((size_t)512*128),256,0,stream>>>(FP(28), attWmT, 128,512,512,0);
  prep_bw_k<<<G((size_t)96*256),256,0,stream>>>(FP(32), preW1T, 256,80,96,0);
  prep_bw_k<<<G((size_t)256*256),256,0,stream>>>(FP(34), preW2T, 256,256,256,0);
  prep_bw_k<<<G((size_t)768*4096),256,0,stream>>>(FP(36), d1WihT, 4096,768,768,1024);
  prep_bw_k<<<G((size_t)1024*4096),256,0,stream>>>(FP(37), d1WhhT, 4096,1024,1024,1024);
  interleave_bias_k<<<G(4096),256,0,stream>>>(FP(38), d1bI, 1024);
  prep_bw_k<<<G((size_t)1536*4096),256,0,stream>>>(FP(39), d2WihT, 4096,1536,1536,1024);
  prep_bw_k<<<G((size_t)1024*4096),256,0,stream>>>(FP(40), d2WhhT, 4096,1024,1024,1024);
  interleave_bias_k<<<G(4096),256,0,stream>>>(FP(41), d2bI, 1024);
  prep_bw_k<<<G((size_t)1536*80),256,0,stream>>>(FP(42), projT, 80,1536,1536,0);
  prep_convw_k<<<G((size_t)5*96*512),256,0,stream>>>(FP(46), pnWT[0], 512,80,5,96);
  for(int i=1;i<4;++i)
    prep_convw_k<<<G((size_t)5*512*512),256,0,stream>>>(FP(46+6*i), pnWT[i], 512,512,5,512);
  prep_convw_k<<<G((size_t)5*512*80),256,0,stream>>>(FP(70), pnWT[4], 80,512,5,512);

  auto gemm = [&](const bf16* A, const bf16* W, const float* bias,
                  const float* g, const float* bb, const float* m, const float* v,
                  bf16* outB, float* outF,
                  int Bsz, int Trows, int TT, int taps, int Cip, int Co,
                  int TpadIn, int tOffIn, int TpadOut, int tOffOut, int ldcF, int act){
    GemmP pp{A,W,bias,g,bb,m,v,outB,outF,Bsz,Trows,TT,taps,Cip,Co,TpadIn,tOffIn,TpadOut,tOffOut,ldcF,act};
    dim3 grid((unsigned)(Bsz*TT), (unsigned)(Co/16));
    gemm_conv_k<<<grid,32,0,stream>>>(pp);
  };

  // ---- encoder ----
  embed_k<<<G((size_t)8*128*512),256,0,stream>>>(tokens, emb, embPad);
  gemm(embPad, encWT[0], FP(4),  FP(5), FP(6), FP(7), FP(8),  cbufA, nullptr, 8,128,8,5,512,512, 132,0,132,2, 0,1);
  gemm(cbufA,  encWT[1], FP(10), FP(11),FP(12),FP(13),FP(14), cbufB, nullptr, 8,128,8,5,512,512, 132,0,132,2, 0,1);
  gemm(cbufB,  encWT[2], FP(16), FP(17),FP(18),FP(19),FP(20), cbufA, nullptr, 8,128,8,5,512,512, 132,0,132,2, 0,1);
  gemm(cbufA, WihT_f, lstmBI_f, nullptr,nullptr,nullptr,nullptr, nullptr, xW_f, 8,128,8,1,512,1024, 132,2,0,0, 1024,0);
  gemm(cbufA, WihT_b, lstmBI_b, nullptr,nullptr,nullptr,nullptr, nullptr, xW_b, 8,128,8,1,512,1024, 132,2,0,0, 1024,0);
  LstmScanP sp{xW_f, xW_b, WhhT_f, WhhT_b, memF, memB};
  lstm_scan_k<<<2,256,0,stream>>>(sp);
  gemm(memB, attWmT, nullptr, nullptr,nullptr,nullptr,nullptr, nullptr, procMem, 8,128,8,1,512,128, 128,0,0,0, 128,0);

  // ---- decoder (persistent, single WGP) ----
  float* out_frames = (float*)d_out;
  float* out_mel2   = (float*)d_out + 128000;
  float* out_stops  = (float*)d_out + 256000;
  float* out_align  = (float*)d_out + 257600;
  DecP dp{ gt, preW1T, FP(33), preW2T, FP(35),
           d1WihT, d1WhhT, d1bI, d2WihT, d2WhhT, d2bI,
           attWqT, FP(30), FP(29), FP(31), procMem, memF, tokens,
           projT, FP(43), FP(44), FP(45),
           out_frames, out_stops, out_align };
  decoder_k<<<1,1024,0,stream>>>(dp);

  // ---- postnet ----
  pn_in_k<<<G((size_t)8*200*96),256,0,stream>>>(out_frames, pnIn);
  gemm(pnIn, pnWT[0], FP(47), FP(48),FP(49),FP(50),FP(51), pnB0, nullptr, 8,200,13,5,96, 512, 212,0,212,2, 0,2);
  gemm(pnB0, pnWT[1], FP(53), FP(54),FP(55),FP(56),FP(57), pnB1, nullptr, 8,200,13,5,512,512, 212,0,212,2, 0,2);
  gemm(pnB1, pnWT[2], FP(59), FP(60),FP(61),FP(62),FP(63), pnB0, nullptr, 8,200,13,5,512,512, 212,0,212,2, 0,2);
  gemm(pnB0, pnWT[3], FP(65), FP(66),FP(67),FP(68),FP(69), pnB1, nullptr, 8,200,13,5,512,512, 212,0,212,2, 0,2);
  gemm(pnB1, pnWT[4], FP(71), FP(72),FP(73),FP(74),FP(75), nullptr, resid, 8,200,13,5,512,80, 212,0,0,0, 80,0);
  combine_k<<<G(128000),256,0,stream>>>(out_frames, resid, out_mel2);
}
